// Circulate_NT_Xent_68152541053482
// MI455X (gfx1250) — matrix operand order repaired and checked by reading, not compiled, not run on hardware
//
#include <hip/hip_runtime.h>
#include <hip/hip_bf16.h>

typedef _Float16 h8   __attribute__((ext_vector_type(8)));
typedef _Float16 h16  __attribute__((ext_vector_type(16)));
typedef float    f8   __attribute__((ext_vector_type(8)));

#define BATCH  4096
#define NROWS  8192
#define DIM    256
#define INV_T  2.0f                       // 1 / TEMPERATURE
#define EXP2_SCALE 2.8853900817779268f    // INV_T * log2(e)
#define MASK_NEG   -200.0f                // exp2(-200) == +0.0f (below denorm range)
#define NCHUNK 8                          // column-sweep split
#define COLS_PER_CHUNK (NROWS / NCHUNK)

#if defined(__has_builtin)
#  if __has_builtin(__builtin_amdgcn_exp2f)
#    define RAW_EXP2(x) __builtin_amdgcn_exp2f(x)
#  else
#    define RAW_EXP2(x) exp2f(x)
#  endif
#else
#  define RAW_EXP2(x) exp2f(x)
#endif

// ---------------------------------------------------------------------------
// Kernel 1: row-normalize z = concat(z_i, z_j) into f16, build concat labels,
// zero the per-row sum-of-exp accumulator. One wave per row, 8 rows/block.
// ---------------------------------------------------------------------------
__global__ __launch_bounds__(256) void ntxent_prep(const float* __restrict__ zi,
                                                   const float* __restrict__ zj,
                                                   const int*   __restrict__ label,
                                                   _Float16*    __restrict__ zn,
                                                   int*         __restrict__ lab2,
                                                   float*       __restrict__ srow)
{
    const int row  = blockIdx.x * 8 + (threadIdx.x >> 5);
    const int lane = threadIdx.x & 31;

    const float* src = (row < BATCH) ? (zi + (size_t)row * DIM)
                                     : (zj + (size_t)(row - BATCH) * DIM);
    float4 a = ((const float4*)src)[2 * lane + 0];
    float4 b = ((const float4*)src)[2 * lane + 1];

    float ss = a.x*a.x + a.y*a.y + a.z*a.z + a.w*a.w
             + b.x*b.x + b.y*b.y + b.z*b.z + b.w*b.w;
#pragma unroll
    for (int m = 16; m >= 1; m >>= 1) ss += __shfl_xor(ss, m, 32);

    const float inv = 1.0f / fmaxf(sqrtf(ss), 1e-8f);

    h8 o;
    o[0] = (_Float16)(a.x * inv); o[1] = (_Float16)(a.y * inv);
    o[2] = (_Float16)(a.z * inv); o[3] = (_Float16)(a.w * inv);
    o[4] = (_Float16)(b.x * inv); o[5] = (_Float16)(b.y * inv);
    o[6] = (_Float16)(b.z * inv); o[7] = (_Float16)(b.w * inv);
    *(h8*)(zn + (size_t)row * DIM + lane * 8) = o;

    if (lane == 0) lab2[row] = label[row & (BATCH - 1)];
    if (lane == 1) srow[row] = 0.0f;
}

// ---------------------------------------------------------------------------
// Kernel 2: positive logits  pos[i] = (zn_i . zn_{i±B}) / T.  One wave/row.
// ---------------------------------------------------------------------------
__global__ __launch_bounds__(256) void ntxent_pos(const _Float16* __restrict__ zn,
                                                  float* __restrict__ posv)
{
    const int row  = blockIdx.x * 8 + (threadIdx.x >> 5);
    const int lane = threadIdx.x & 31;
    const int prt  = (row < BATCH) ? (row + BATCH) : (row - BATCH);

    h8 a = *(const h8*)(zn + (size_t)row * DIM + lane * 8);
    h8 b = *(const h8*)(zn + (size_t)prt * DIM + lane * 8);
    float d = 0.0f;
#pragma unroll
    for (int e = 0; e < 8; ++e) d += (float)a[e] * (float)b[e];
#pragma unroll
    for (int m = 16; m >= 1; m >>= 1) d += __shfl_xor(d, m, 32);
    if (lane == 0) posv[row] = d * INV_T;
}

// ---------------------------------------------------------------------------
// Kernel 3: fused sim tile + masked sum-of-exp.  One 16-row tile per wave,
// one column chunk per blockIdx.y.  sim/T in [-2,2] => no running max needed.
// Mask is folded into the exp argument (-200 -> exp2 == +0), so the
// accumulate is a plain packed add and the exp is a bare v_exp_f32.
// ---------------------------------------------------------------------------
__global__ __launch_bounds__(128) void ntxent_main(const _Float16* __restrict__ zn,
                                                   const int*      __restrict__ lab2,
                                                   float*          __restrict__ srow)
{
    const int lane = threadIdx.x & 31;
    const int wave = threadIdx.x >> 5;
    const int i0   = (blockIdx.x * 4 + wave) * 16;     // 16-row tile
    const int c0   = blockIdx.y * COLS_PER_CHUNK;      // column chunk
    const int half = lane >> 4;
    const int lc   = lane & 15;

    // A fragments: 16 rows x 256 K in registers (contiguous k = 16*half + s, matching B).
    h16 afrag[8];
    {
        const _Float16* arow = zn + (size_t)(i0 + lc) * DIM;
#pragma unroll
        for (int kk = 0; kk < 8; ++kk) {
            h8 lo = *(const h8*)(arow + kk * 32 + half * 16);
            h8 hi = *(const h8*)(arow + kk * 32 + half * 16 + 8);
            h16 f;
#pragma unroll
            for (int e = 0; e < 8; ++e) { f[e] = lo[e]; f[e + 8] = hi[e]; }
            afrag[kk] = f;
        }
    }

    int labi[8];
#pragma unroll
    for (int r = 0; r < 8; ++r) labi[r] = lab2[i0 + half * 8 + r];

    float s[8];
#pragma unroll
    for (int r = 0; r < 8; ++r) s[r] = 0.0f;

    for (int j0 = c0; j0 < c0 + COLS_PER_CHUNK; j0 += 16) {
        const int j    = j0 + lc;
        const int labj = lab2[j];
        const _Float16* brow = zn + (size_t)j * DIM + half * 16;

        f8 acc = {};
#pragma unroll
        for (int kk = 0; kk < 8; ++kk) {
            h16 b = *(const h16*)(brow + kk * 32);
            acc = __builtin_amdgcn_wmma_f32_16x16x32_f16(
                      false, afrag[kk], false, b, (short)0, acc, false, false);
        }

#pragma unroll
        for (int r = 0; r < 8; ++r) {
            float xs = acc[r] * EXP2_SCALE;              // log2(exp(sim/T))
            xs = (labj == labi[r]) ? MASK_NEG : xs;      // mask -> exp2 == 0
            s[r] += RAW_EXP2(xs);
        }
    }

    // reduce across the 16 lanes of each half-wave, then combine chunks.
#pragma unroll
    for (int r = 0; r < 8; ++r) {
#pragma unroll
        for (int m = 1; m <= 8; m <<= 1) s[r] += __shfl_xor(s[r], m, 32);
    }
    if (lc == 0) {
#pragma unroll
        for (int r = 0; r < 8; ++r) atomicAdd(&srow[i0 + half * 8 + r], s[r]);
    }
}

// ---------------------------------------------------------------------------
// Kernel 4: single-block finalization.  hist -> counts, per-row loss, reduce.
// ---------------------------------------------------------------------------
__global__ __launch_bounds__(1024) void ntxent_final(const float* __restrict__ srow,
                                                     const float* __restrict__ posv,
                                                     const int*   __restrict__ label,
                                                     float*       __restrict__ out)
{
    __shared__ int   hist[16];
    __shared__ float wsum[32];
    const int t = threadIdx.x;

    if (t < 16) hist[t] = 0;
    __syncthreads();
    for (int i = t; i < BATCH; i += 1024) atomicAdd(&hist[label[i]], 1);
    __syncthreads();

    float tot = 0.0f;
    for (int i = t; i < NROWS; i += 1024) {
        const float p   = posv[i];
        const float cnt = 2.0f * (float)hist[label[i & (BATCH - 1)]];
        tot += (__logf(srow[i] + __expf(p)) - p) / cnt;
    }
#pragma unroll
    for (int m = 16; m >= 1; m >>= 1) tot += __shfl_xor(tot, m, 32);
    if ((t & 31) == 0) wsum[t >> 5] = tot;
    __syncthreads();
    if (t < 32) {
        float v = wsum[t];
#pragma unroll
        for (int m = 16; m >= 1; m >>= 1) v += __shfl_xor(v, m, 32);
        if (t == 0) out[0] = v;
    }
}

// ---------------------------------------------------------------------------
extern "C" void kernel_launch(void* const* d_in, const int* in_sizes, int n_in,
                              void* d_out, int out_size, void* d_ws, size_t ws_size,
                              hipStream_t stream)
{
    const float* zi    = (const float*)d_in[0];
    const float* zj    = (const float*)d_in[1];
    const int*   label = (const int*)d_in[2];
    float*       out   = (float*)d_out;

    char* ws = (char*)d_ws;
    _Float16* zn   = (_Float16*)ws;                                  // 4 MB
    int*      lab2 = (int*)  (ws + (size_t)NROWS * DIM * 2);         // 32 KB
    float*    srow = (float*)(ws + (size_t)NROWS * DIM * 2 + 32768); // 32 KB
    float*    posv = (float*)(ws + (size_t)NROWS * DIM * 2 + 65536); // 32 KB

    ntxent_prep<<<NROWS / 8, 256, 0, stream>>>(zi, zj, label, zn, lab2, srow);
    ntxent_pos <<<NROWS / 8, 256, 0, stream>>>(zn, posv);
    ntxent_main<<<dim3(NROWS / 16 / 4, NCHUNK), 128, 0, stream>>>(zn, lab2, srow);
    ntxent_final<<<1, 1024, 0, stream>>>(srow, posv, label, out);
}
